// MPA_16698832847132
// MI455X (gfx1250) — matrix operand order repaired and checked by reading, not compiled, not run on hardware
//
#include <hip/hip_runtime.h>

typedef __attribute__((ext_vector_type(16))) __bf16 v16bf;
typedef __attribute__((ext_vector_type(8)))  float  v8f;

// ---------- bf16 helpers (RNE) ----------
__device__ __forceinline__ unsigned short f2bf(float f) {
  unsigned int u = __float_as_uint(f);
  u += 0x7FFFu + ((u >> 16) & 1u);
  return (unsigned short)(u >> 16);
}
__device__ __forceinline__ float bf2f(unsigned short h) {
  return __uint_as_float(((unsigned int)h) << 16);
}

union VB { v16bf v; unsigned int u[8]; };

// ---------- WMMA fragment loaders (CDNA5 wave32 layouts, CDNA5 ISA §7.12.2) ----------
// A 16x32 bf16: lane holds row M=lane%16; VGPR v -> K = (v<4 ? 2v : 8+2v) + (lane>=16 ? 8 : 0)
__device__ __forceinline__ v16bf load_a16x32(const unsigned short* base, int ldk, int lane) {
  VB r;
  const unsigned short* row = base + (lane & 15) * ldk;
  const int koff = (lane & 16) ? 8 : 0;
#pragma unroll
  for (int i = 0; i < 8; ++i) {
    int k = ((i < 4) ? (2 * i) : (8 + 2 * i)) + koff;
    r.u[i] = *(const unsigned int*)(row + k);
  }
  return r.v;
}
// B 32x16 bf16 from Bt stored [N][K] (K contiguous): lane n=lane%16, K = (v<4 ? 2v : 8+2v) + (lane>=16 ? 8 : 0)
__device__ __forceinline__ v16bf load_b32x16(const unsigned short* bt, int ldk, int lane) {
  VB r;
  const unsigned short* row = bt + (lane & 15) * ldk + ((lane & 16) ? 8 : 0);
#pragma unroll
  for (int i = 0; i < 8; ++i) r.u[i] = *(const unsigned int*)(row + ((i < 4) ? (2 * i) : (8 + 2 * i)));
  return r.v;
}
// A loader for implicit-GEMM conv: rows are 2D pixels of an outW-wide output grid,
// input stored [y*inW+x][32ch] bf16 in LDS; one WMMA K-chunk = 32 channels of one tap.
__device__ __forceinline__ v16bf load_a_conv(const unsigned short* in, int p0, int outW, int inW,
                                             int ky, int kx, int maxP, int lane) {
  VB r;
  int p = p0 + (lane & 15);
  if (p > maxP) p = maxP;
  int y = p / outW, x = p - y * outW;
  const unsigned short* row = in + ((y + ky) * inW + (x + kx)) * 32;
  const int koff = (lane & 16) ? 8 : 0;
#pragma unroll
  for (int i = 0; i < 8; ++i) {
    int k = ((i < 4) ? (2 * i) : (8 + 2 * i)) + koff;
    r.u[i] = *(const unsigned int*)(row + k);
  }
  return r.v;
}

__device__ __forceinline__ v8f wmma_bf16(v16bf a, v16bf b, v8f c) {
  return __builtin_amdgcn_wmma_f32_16x16x32_bf16(false, a, false, b, (short)0, c, false, false);
}
__device__ __forceinline__ v8f zero8() {
  v8f z = {0.f, 0.f, 0.f, 0.f, 0.f, 0.f, 0.f, 0.f};
  return z;
}
// reductions across the 16-lane half-group (rows of C fragments live across 16 lanes)
__device__ __forceinline__ float rmax16(float v) {
  v = fmaxf(v, __shfl_xor(v, 1));
  v = fmaxf(v, __shfl_xor(v, 2));
  v = fmaxf(v, __shfl_xor(v, 4));
  v = fmaxf(v, __shfl_xor(v, 8));
  return v;
}
__device__ __forceinline__ float rsum16(float v) {
  v += __shfl_xor(v, 1);
  v += __shfl_xor(v, 2);
  v += __shfl_xor(v, 4);
  v += __shfl_xor(v, 8);
  return v;
}

// ---------- fp32 -> bf16 convert ----------
__global__ void cvt_bf16(const float* __restrict__ src, unsigned short* __restrict__ dst, int n) {
  for (int i = blockIdx.x * blockDim.x + threadIdx.x; i < n; i += gridDim.x * blockDim.x)
    dst[i] = f2bf(src[i]);
}

// ---------- QKV GEMM: qkv = x(8192x512) @ wqkv^T(512x1536), scatter to per-head layout ----------
// wave tile: 16 rows x 128 cols (8 C fragments) -> A fragment reused 8x per load
__global__ void __launch_bounds__(256)
qkv_gemm(const unsigned short* __restrict__ xb, const unsigned short* __restrict__ wb,
         unsigned short* __restrict__ qh, unsigned short* __restrict__ kh,
         unsigned short* __restrict__ vh) {
  const int lane = threadIdx.x & 31, wave = threadIdx.x >> 5;
  const int m0 = blockIdx.x * 128 + wave * 16;
  const int n0 = blockIdx.y * 128;
  v8f c[8];
#pragma unroll
  for (int t = 0; t < 8; ++t) c[t] = zero8();
  for (int k0 = 0; k0 < 512; k0 += 32) {
    v16bf a = load_a16x32(xb + (size_t)m0 * 512 + k0, 512, lane);
#pragma unroll
    for (int t = 0; t < 8; ++t)
      c[t] = wmma_bf16(a, load_b32x16(wb + (size_t)(n0 + t * 16) * 512 + k0, 512, lane), c[t]);
  }
  const int hl = lane >> 4, col = lane & 15;
#pragma unroll
  for (int t = 0; t < 8; ++t) {
    int n = n0 + t * 16 + col;
    int sec = n >> 9, hn = n & 511, h = hn >> 6, d = hn & 63;
    unsigned short* dst = (sec == 0) ? qh : (sec == 1 ? kh : vh);
#pragma unroll
    for (int r = 0; r < 8; ++r) {
      int m = m0 + r + 8 * hl;
      int bb = m >> 10, nr = m & 1023;
      dst[(((size_t)(bb * 8 + h) * 1024 + nr) << 6) + d] = f2bf(c[t][r]);
    }
  }
}

// ---------- PPE: per-image conv->GN->ReLU->conv->GN->sigmoid, WMMA implicit-GEMM convs ----------
__global__ void __launch_bounds__(256)
ppe_kernel(const float* __restrict__ rpe,
           const float* __restrict__ c1w, const float* __restrict__ c1b,
           const float* __restrict__ g1w, const float* __restrict__ g1b,
           const float* __restrict__ c2w, const float* __restrict__ c2b,
           const float* __restrict__ g2w, const float* __restrict__ g2b,
           float* __restrict__ wbias) {
  extern __shared__ unsigned char smem[];
  unsigned short* inT = (unsigned short*)smem;              // [36*36][32] bf16  (82944 B)
  unsigned short* c1o = (unsigned short*)(smem + 82944);    // [34*34][32] bf16  (73984 B)
  unsigned short* w1t = (unsigned short*)(smem + 156928);   // [32 oc][9*32]     (18432 B)
  unsigned short* w2t = (unsigned short*)(smem + 175360);   // [16 oc][9*32]     ( 9216 B)
  float* c2o = (float*)(smem + 184576);                     // [8][1024] f32     (32768 B)
  float* stat = (float*)(smem + 217344);                    // 160 f32
  float* sumA = stat;        float* sqA = stat + 32;
  float* scA  = stat + 64;   float* shA = stat + 96;
  float* sumB = stat + 128;  float* sqB = stat + 136;
  float* scB  = stat + 144;  float* shB = stat + 152;

  const int tid = threadIdx.x, lane = tid & 31, wave = tid >> 5;
  const int n = blockIdx.x, ii = n >> 5, jj = n & 31;

  if (tid < 160) stat[tid] = 0.f;

  // load the rpe row for this image (contiguous 1296*32 floats) as bf16
  const float* src = rpe + (size_t)((ii + 2) * 36 + (jj + 2)) * (1296 * 32);
  for (int e = tid; e < 1296 * 32; e += 256) inT[e] = f2bf(src[e]);
  // rearrange weights: w1t[oc][tap*32+ic], w2t padded to 16 output channels
  for (int e = tid; e < 32 * 288; e += 256) {
    int oc = e / 288, r = e - oc * 288, tap = r >> 5, ic = r & 31;
    w1t[e] = f2bf(c1w[(oc * 32 + ic) * 9 + tap]);
  }
  for (int e = tid; e < 16 * 288; e += 256) {
    int oc = e / 288, r = e - oc * 288, tap = r >> 5, ic = r & 31;
    w2t[e] = f2bf(oc < 8 ? c2w[(oc * 32 + ic) * 9 + tap] : 0.f);
  }
  __syncthreads();

  const int hl = lane >> 4, col = lane & 15;
  // conv1: M=1156 pixels (34x34), N=32, K=288; one WMMA per tap per 16-oc tile
  for (int tile = wave; tile < 73; tile += 8) {
    int p0 = tile * 16;
    v8f c0 = zero8(), c1 = zero8();
#pragma unroll
    for (int tap = 0; tap < 9; ++tap) {
      int ky = tap / 3, kx = tap - ky * 3;
      v16bf a = load_a_conv(inT, p0, 34, 36, ky, kx, 1155, lane);
      c0 = wmma_bf16(a, load_b32x16(w1t + tap * 32, 288, lane), c0);
      c1 = wmma_bf16(a, load_b32x16(w1t + 16 * 288 + tap * 32, 288, lane), c1);
    }
#pragma unroll
    for (int r = 0; r < 8; ++r) {
      int p = p0 + r + 8 * hl;
      if (p < 1156) {
        c1o[p * 32 + col] = f2bf(c0[r] + c1b[col]);
        c1o[p * 32 + 16 + col] = f2bf(c1[r] + c1b[16 + col]);
      }
    }
  }
  __syncthreads();
  // GroupNorm 1 (per-channel over 1156 pixels)
  {
    int oc = tid & 31, sub = tid >> 5;
    float s = 0.f, q = 0.f;
    for (int p = sub; p < 1156; p += 8) {
      float v = bf2f(c1o[p * 32 + oc]);
      s += v; q += v * v;
    }
    atomicAdd(&sumA[oc], s);
    atomicAdd(&sqA[oc], q);
  }
  __syncthreads();
  if (tid < 32) {
    float mu = sumA[tid] * (1.f / 1156.f);
    float var = sqA[tid] * (1.f / 1156.f) - mu * mu;
    float rs = rsqrtf(var + 1e-5f);
    scA[tid] = g1w[tid] * rs;
    shA[tid] = g1b[tid] - mu * g1w[tid] * rs;
  }
  __syncthreads();
  for (int e = tid; e < 1156 * 32; e += 256) {
    int oc = e & 31;
    float v = bf2f(c1o[e]) * scA[oc] + shA[oc];
    c1o[e] = f2bf(fmaxf(v, 0.f));
  }
  __syncthreads();
  // conv2: M=1024 pixels (32x32), N=8 (padded 16), K=288
  for (int tile = wave; tile < 64; tile += 8) {
    int p0 = tile * 16;
    v8f c = zero8();
#pragma unroll
    for (int tap = 0; tap < 9; ++tap) {
      int ky = tap / 3, kx = tap - ky * 3;
      v16bf a = load_a_conv(c1o, p0, 32, 34, ky, kx, 1023, lane);
      c = wmma_bf16(a, load_b32x16(w2t + tap * 32, 288, lane), c);
    }
    if (col < 8) {
#pragma unroll
      for (int r = 0; r < 8; ++r) {
        int p = p0 + r + 8 * hl;
        c2o[col * 1024 + p] = c[r] + c2b[col];
      }
    }
  }
  __syncthreads();
  // GroupNorm 2 + sigmoid + store w[h][n][m]
  {
    int oc = tid >> 5, sub = tid & 31;
    float s = 0.f, q = 0.f;
    for (int p = sub; p < 1024; p += 32) {
      float v = c2o[oc * 1024 + p];
      s += v; q += v * v;
    }
    atomicAdd(&sumB[oc], s);
    atomicAdd(&sqB[oc], q);
  }
  __syncthreads();
  if (tid < 8) {
    float mu = sumB[tid] * (1.f / 1024.f);
    float var = sqB[tid] * (1.f / 1024.f) - mu * mu;
    float rs = rsqrtf(var + 1e-5f);
    scB[tid] = g2w[tid] * rs;
    shB[tid] = g2b[tid] - mu * g2w[tid] * rs;
  }
  __syncthreads();
  {
    int oc = tid >> 5, sub = tid & 31;
    for (int p = sub; p < 1024; p += 32) {
      float v = c2o[oc * 1024 + p] * scB[oc] + shB[oc];
      wbias[((size_t)oc * 1024 + n) * 1024 + p] = 1.f / (1.f + __expf(-v));
    }
  }
}

// ---------- fused biased attention (flash-style, WMMA) ----------
__global__ void __launch_bounds__(256)
attn_kernel(const unsigned short* __restrict__ qh, const unsigned short* __restrict__ kh,
            const unsigned short* __restrict__ vh, const float* __restrict__ wbias,
            unsigned short* __restrict__ attno) {
  extern __shared__ unsigned char smem[];
  unsigned short* Ks = (unsigned short*)smem;              // [1024][64] at dynamic-LDS offset 0
  unsigned short* Vt = (unsigned short*)(smem + 131072);   // [64][1024]
  unsigned short* Ps = (unsigned short*)(smem + 262144);   // [8 waves][16][32]
  const int tid = threadIdx.x, lane = tid & 31, wave = tid >> 5;
  const int bh = blockIdx.y, b = bh >> 3, h = bh & 7;
  const unsigned short* kb = kh + (size_t)bh * 65536;
  const unsigned short* vb = vh + (size_t)bh * 65536;
  // K tile: 128 KB staged via CDNA5 async copy (ASYNCcnt path, no VGPR bounce).
  // No static __shared__ in this kernel -> dynamic LDS starts at offset 0,
  // so the LDS byte offset of Ks[i*8] is exactly i*16.
  for (int i = tid; i < 8192; i += 256) {
    asm volatile("global_load_async_to_lds_b128 %0, %1, off"
                 :: "v"(i * 16), "v"(kb + (size_t)i * 8)
                 : "memory");
  }
  // V transposed into [d][n] with plain stores (async path cannot transpose)
  for (int e = tid; e < 65536; e += 256) {
    int nn = e >> 6, d = e & 63;
    Vt[d * 1024 + nn] = vb[e];
  }
  asm volatile("s_wait_asynccnt 0x0" ::: "memory");
  __syncthreads();

  const int m0 = blockIdx.x * 128 + wave * 16;
  const int hl = lane >> 4, col = lane & 15;
  const unsigned short* qrow = qh + ((size_t)bh * 1024 + m0) * 64;
  v16bf a0 = load_a16x32(qrow, 64, lane);
  v16bf a1 = load_a16x32(qrow + 32, 64, lane);

  v8f o0 = zero8(), o1 = zero8(), o2 = zero8(), o3 = zero8();
  float mrow[8], lrow[8];
#pragma unroll
  for (int r = 0; r < 8; ++r) { mrow[r] = -3.0e38f; lrow[r] = 0.f; }
  const float scale = 0.125f;  // 64^-0.5
  unsigned short* pw = Ps + wave * 512;
  const float* wbase = wbias + (size_t)(h * 1024 + m0) * 1024;

  for (int j0 = 0; j0 < 1024; j0 += 32) {
    v8f s0, s1;
    s0 = wmma_bf16(a0, load_b32x16(Ks + j0 * 64, 64, lane), zero8());
    s0 = wmma_bf16(a1, load_b32x16(Ks + j0 * 64 + 32, 64, lane), s0);
    s1 = wmma_bf16(a0, load_b32x16(Ks + (j0 + 16) * 64, 64, lane), zero8());
    s1 = wmma_bf16(a1, load_b32x16(Ks + (j0 + 16) * 64 + 32, 64, lane), s1);

    // prefetch next j-step of the bias rows into cache (global_prefetch_b8)
    if (j0 + 32 < 1024)
      __builtin_prefetch(wbase + (size_t)(8 * hl) * 1024 + j0 + 32 + col, 0, 0);

    float alpha[8];
#pragma unroll
    for (int r = 0; r < 8; ++r) {
      float x0 = s0[r] * scale, x1 = s1[r] * scale;
      float mn = fmaxf(mrow[r], rmax16(fmaxf(x0, x1)));
      alpha[r] = __expf(mrow[r] - mn);
      mrow[r] = mn;
      const float* wrow = wbase + (size_t)(r + 8 * hl) * 1024 + j0;
      float p0 = wrow[col] * __expf(x0 - mn);
      float p1 = wrow[16 + col] * __expf(x1 - mn);
      lrow[r] = lrow[r] * alpha[r] + rsum16(p0 + p1);
      unsigned short* pr = pw + (r + 8 * hl) * 32;
      pr[col] = f2bf(p0);
      pr[16 + col] = f2bf(p1);
    }
#pragma unroll
    for (int r = 0; r < 8; ++r) {
      o0[r] *= alpha[r]; o1[r] *= alpha[r]; o2[r] *= alpha[r]; o3[r] *= alpha[r];
    }
    v16bf ap = load_a16x32(pw, 32, lane);  // C-layout -> A-layout via per-wave LDS round trip
    o0 = wmma_bf16(ap, load_b32x16(Vt + 0 * 1024 + j0, 1024, lane), o0);
    o1 = wmma_bf16(ap, load_b32x16(Vt + 16 * 1024 + j0, 1024, lane), o1);
    o2 = wmma_bf16(ap, load_b32x16(Vt + 32 * 1024 + j0, 1024, lane), o2);
    o3 = wmma_bf16(ap, load_b32x16(Vt + 48 * 1024 + j0, 1024, lane), o3);
  }
#pragma unroll
  for (int r = 0; r < 8; ++r) {
    float inv = 1.f / fmaxf(lrow[r], 1e-12f);
    int row = m0 + r + 8 * hl;
    size_t base = ((size_t)(b * 1024 + row) * 512) + h * 64;
    attno[base + col] = f2bf(o0[r] * inv);
    attno[base + 16 + col] = f2bf(o1[r] * inv);
    attno[base + 32 + col] = f2bf(o2[r] * inv);
    attno[base + 48 + col] = f2bf(o3[r] * inv);
  }
}

// ---------- output projection: out = attno(8192x512) @ wproj^T + bproj ----------
__global__ void __launch_bounds__(256)
proj_gemm(const unsigned short* __restrict__ ab, const unsigned short* __restrict__ wb,
          const float* __restrict__ bias, float* __restrict__ out) {
  const int lane = threadIdx.x & 31, wave = threadIdx.x >> 5;
  const int m0 = blockIdx.x * 128 + wave * 16;
  const int n0 = blockIdx.y * 128;
  v8f c[8];
#pragma unroll
  for (int t = 0; t < 8; ++t) c[t] = zero8();
  for (int k0 = 0; k0 < 512; k0 += 32) {
    v16bf a = load_a16x32(ab + (size_t)m0 * 512 + k0, 512, lane);
#pragma unroll
    for (int t = 0; t < 8; ++t)
      c[t] = wmma_bf16(a, load_b32x16(wb + (size_t)(n0 + t * 16) * 512 + k0, 512, lane), c[t]);
  }
  const int hl = lane >> 4, col = lane & 15;
#pragma unroll
  for (int t = 0; t < 8; ++t) {
    int nn = n0 + t * 16 + col;
    float bv = bias[nn];
#pragma unroll
    for (int r = 0; r < 8; ++r) {
      int m = m0 + r + 8 * hl;
      out[(size_t)m * 512 + nn] = c[t][r] + bv;
    }
  }
}

extern "C" void kernel_launch(void* const* d_in, const int* in_sizes, int n_in,
                              void* d_out, int out_size, void* d_ws, size_t ws_size,
                              hipStream_t stream) {
  const float* x = (const float*)d_in[0];
  const float* rpe = (const float*)d_in[1];
  const float* wqkv = (const float*)d_in[2];
  const float* wproj = (const float*)d_in[3];
  const float* bproj = (const float*)d_in[4];
  const float* c1w = (const float*)d_in[5];
  const float* c1b = (const float*)d_in[6];
  const float* g1w = (const float*)d_in[7];
  const float* g1b = (const float*)d_in[8];
  const float* c2w = (const float*)d_in[9];
  const float* c2b = (const float*)d_in[10];
  const float* g2w = (const float*)d_in[11];
  const float* g2b = (const float*)d_in[12];

  unsigned char* ws = (unsigned char*)d_ws;
  unsigned short* xb    = (unsigned short*)(ws + 0);         //  8 MB
  unsigned short* wqb   = (unsigned short*)(ws + 8388608);   //  1.5 MB
  unsigned short* wpb   = (unsigned short*)(ws + 9961472);   //  0.5 MB
  unsigned short* qh    = (unsigned short*)(ws + 10485760);  //  8 MB
  unsigned short* kh    = (unsigned short*)(ws + 18874368);  //  8 MB
  unsigned short* vh    = (unsigned short*)(ws + 27262976);  //  8 MB
  float*          wbias = (float*)(ws + 35651584);           // 32 MB
  unsigned short* ao    = (unsigned short*)(ws + 69206016);  //  8 MB

  cvt_bf16<<<4096, 256, 0, stream>>>(x, xb, 8 * 1024 * 512);
  cvt_bf16<<<1024, 256, 0, stream>>>(wqkv, wqb, 1536 * 512);
  cvt_bf16<<<512, 256, 0, stream>>>(wproj, wpb, 512 * 512);

  qkv_gemm<<<dim3(64, 12), 256, 0, stream>>>(xb, wqb, qh, kh, vh);
  ppe_kernel<<<1024, 256, 217984, stream>>>(rpe, c1w, c1b, g1w, g1b,
                                            c2w, c2b, g2w, g2b, wbias);
  attn_kernel<<<dim3(8, 64), 256, 270336, stream>>>(qh, kh, vh, wbias, ao);
  proj_gemm<<<dim3(64, 4), 256, 0, stream>>>(ao, wpb, bproj, (float*)d_out);
}
